// DMPNNEncoder_65558380806592
// MI455X (gfx1250) — matrix operand order repaired and checked by reading, not compiled, not run on hardware
//
#include <hip/hip_runtime.h>

// ---------------------------------------------------------------------------
// DMPNN encoder for gfx1250 (MI455X): bf16 WMMA GEMMs with fused gather /
// scatter epilogues. wave32; v_wmma_f32_16x16x32_bf16 is the compute core.
// Each wave computes a 32x64 output block (2 M-tiles x 4 N-tiles) so every
// B fragment load feeds two WMMAs; all loop trip counts are compile-time and
// all fragment loads lower to base + immediate-offset b128 clauses.
// ---------------------------------------------------------------------------

#define NUM_ATOMS  131072
#define NUM_EDGES  262144
#define NUM_GRAPHS 4096
#define ATOM_FDIM  133
#define BOND_FDIM  14
#define HIDDEN     300

#define NT         20          // N tiles of 16 covering 320 (>=300), 5 blocks x 4 tiles
#define JT         4           // N tiles per wave (compile-time)
#define MT         2           // M tiles per wave
#define GRIDY      5
#define MSG_STRIDE 320         // messages row stride (f32)
#define NEI_STRIDE 320         // nei row stride (f32), K padded to 32
#define KA_INIT    160         // 147 -> pad 160  (5 chunks of 32)
#define KC_INIT    5
#define KA_UPD     320         // 300 -> pad 320  (10 chunks)
#define KC_UPD     10
#define KA_OUT     448         // 433 -> pad 448  (14 chunks)
#define KC_OUT     14

typedef __bf16 bf16;
typedef __attribute__((ext_vector_type(16))) __bf16 v16bf;
typedef __attribute__((ext_vector_type(8)))  __bf16 v8bf;
typedef __attribute__((ext_vector_type(8)))  float  v8f;

// ------------------------- helper kernels ----------------------------------

__global__ void zero_f32(float* __restrict__ p, int n) {
    int i = blockIdx.x * blockDim.x + threadIdx.x;
    if (i < n) p[i] = 0.0f;
}

// Pack weight W[Kreal x 300] (row-major, y = x @ W) into WMMA B fragments.
// Fragment (kc, nt) holds B tile K=[kc*32,kc*32+32) x N=[nt*16,nt*16+16):
//   element (lane, i):  K = kc*32 + (i<8?0:16) + (lane<16?0:8) + (i&7),  N = nt*16 + lane%16
// stored at dst[(kc*NT+nt)*512 + lane*16 + i]  -> one 32B load per lane.
__global__ void prep_wfrag(const float* __restrict__ W, bf16* __restrict__ dst,
                           int Kreal, int KC) {
    int total = KC * NT * 512;
    int tid = blockIdx.x * blockDim.x + threadIdx.x;
    if (tid >= total) return;
    int f    = tid >> 9;
    int r    = tid & 511;
    int lane = r >> 4;
    int i    = r & 15;
    int kc = f / NT, nt = f % NT;
    int k = kc * 32 + ((i < 8) ? 0 : 16) + ((lane < 16) ? 0 : 8) + (i & 7);
    int n = nt * 16 + (lane & 15);
    float v = (k < Kreal && n < HIDDEN) ? W[k * HIDDEN + n] : 0.0f;
    dst[tid] = (bf16)v;
}

// A_init[e][k] = bf16( k<133 ? x[src[e]][k] : k<147 ? edge_attr[e][k-133] : 0 )
__global__ void build_a_init(const float* __restrict__ x,
                             const float* __restrict__ ea,
                             const int* __restrict__ src,
                             bf16* __restrict__ A) {
    int tid = blockIdx.x * blockDim.x + threadIdx.x;   // E*160 threads
    int e = tid / KA_INIT;
    int k = tid - e * KA_INIT;
    if (e >= NUM_EDGES) return;
    float v = 0.0f;
    if (k < ATOM_FDIM)                  v = x[(size_t)src[e] * ATOM_FDIM + k];
    else if (k < ATOM_FDIM + BOND_FDIM) v = ea[(size_t)e * BOND_FDIM + (k - ATOM_FDIM)];
    A[tid] = (bf16)v;
}

// A_atom[row][k] = bf16( k<133 ? x[row][k] : k<433 ? atom_msg[row][k-133] : 0 )
__global__ void build_a_atom(const float* __restrict__ x,
                             const float* __restrict__ am,   // [N, 320]
                             bf16* __restrict__ A) {
    int tid = blockIdx.x * blockDim.x + threadIdx.x;   // N*448 threads
    int row = tid / KA_OUT;
    int k   = tid - row * KA_OUT;
    if (row >= NUM_ATOMS) return;
    float v = 0.0f;
    if (k < ATOM_FDIM)               v = x[(size_t)row * ATOM_FDIM + k];
    else if (k < ATOM_FDIM + HIDDEN) v = am[(size_t)row * NEI_STRIDE + (k - ATOM_FDIM)];
    A[tid] = (bf16)v;
}

// ------------------------- WMMA fragment loads -----------------------------

// A 16x32 bf16 fragment from a row-major bf16 matrix.
// lane: M = lane%16; element i<8 -> K = kbase + (lane<16?0:8) + i
//                    element i>=8 -> K = kbase + 16 + (lane<16?0:8) + (i-8)
__device__ inline v16bf a_frag_bf16(const bf16* __restrict__ row_ptr, int kbase) {
    v8bf lo = *(const v8bf*)(row_ptr + kbase);
    v8bf hi = *(const v8bf*)(row_ptr + kbase + 16);
    v16bf a;
#pragma unroll
    for (int i = 0; i < 8; ++i) { a[i] = lo[i]; a[8 + i] = hi[i]; }
    return a;
}

// Same fragment gathered from a row-major f32 matrix (convert in-register).
__device__ inline v16bf a_frag_f32(const float* __restrict__ row_ptr, int kbase) {
    const float* p = row_ptr + kbase;
    float4 l0 = *(const float4*)(p);
    float4 l1 = *(const float4*)(p + 4);
    float4 h0 = *(const float4*)(p + 16);
    float4 h1 = *(const float4*)(p + 20);
    v16bf a;
    a[0]=(bf16)l0.x;  a[1]=(bf16)l0.y;  a[2]=(bf16)l0.z;  a[3]=(bf16)l0.w;
    a[4]=(bf16)l1.x;  a[5]=(bf16)l1.y;  a[6]=(bf16)l1.z;  a[7]=(bf16)l1.w;
    a[8]=(bf16)h0.x;  a[9]=(bf16)h0.y;  a[10]=(bf16)h0.z; a[11]=(bf16)h0.w;
    a[12]=(bf16)h1.x; a[13]=(bf16)h1.y; a[14]=(bf16)h1.z; a[15]=(bf16)h1.w;
    return a;
}

// B fragment: per-lane base + compile-time element offset -> immediate ioffset.
__device__ inline v16bf b_frag(const bf16* __restrict__ bbase, int kc, int j) {
    return *(const v16bf*)(bbase + (size_t)(kc * NT + 5 * j) * 512);
}

__device__ inline float relu(float v) { return v > 0.0f ? v : 0.0f; }

// ------------------------- GEMM 1: init messages ---------------------------
// messages = relu(A_init @ W_i);  nei += scatter_by_tgt(messages)
__global__ __launch_bounds__(128) void gemm_msg_init(
    const bf16* __restrict__ A,       // [E, 160] bf16
    const bf16* __restrict__ Wfrag,   // KC_INIT*NT fragments
    const int*  __restrict__ tgt,
    float* __restrict__ messages,     // [E, 320]
    float* __restrict__ nei)          // [N, 320] pre-zeroed
{
    const int lane  = threadIdx.x & 31;
    const int wave  = threadIdx.x >> 5;
    const int mrow0 = (blockIdx.x * 4 + wave) * (16 * MT);   // first of 32 rows
    const int ntSel = blockIdx.y;                            // 0..4
    const int khalf = (lane < 16) ? 0 : 8;

    const bf16* arow0 = A + (size_t)(mrow0 + (lane & 15)) * KA_INIT + khalf;
    const bf16* arow1 = arow0 + (size_t)16 * KA_INIT;
    const bf16* bbase = Wfrag + (size_t)ntSel * 512 + lane * 16;

    v8f acc[MT][JT] = {};
#pragma unroll
    for (int kc = 0; kc < KC_INIT; ++kc) {
        v16bf a0 = a_frag_bf16(arow0, kc * 32);
        v16bf a1 = a_frag_bf16(arow1, kc * 32);
#pragma unroll
        for (int j = 0; j < JT; ++j) {
            v16bf b = b_frag(bbase, kc, j);
            acc[0][j] = __builtin_amdgcn_wmma_f32_16x16x32_bf16(
                false, a0, false, b, (short)0, acc[0][j], false, false);
            acc[1][j] = __builtin_amdgcn_wmma_f32_16x16x32_bf16(
                false, a1, false, b, (short)0, acc[1][j], false, false);
        }
    }

    const int ncol  = lane & 15;
    const int nbase = ntSel * 16 + ncol;
#pragma unroll
    for (int m = 0; m < MT; ++m) {
        const int rbase = mrow0 + m * 16 + ((lane < 16) ? 0 : 8);
        float* mbase = messages + (size_t)rbase * MSG_STRIDE + nbase;
        int t[8];
#pragma unroll
        for (int r = 0; r < 8; ++r) t[r] = tgt[rbase + r];
#pragma unroll
        for (int j = 0; j < JT; ++j) {
            int n = nbase + 80 * j;
#pragma unroll
            for (int r = 0; r < 8; ++r) {
                float v = relu(acc[m][j][r]);
                mbase[(size_t)r * MSG_STRIDE + 80 * j] = v;
                if (n < HIDDEN) atomicAdd(nei + (size_t)t[r] * NEI_STRIDE + n, v);
            }
        }
    }
}

// ------------------------- GEMM 2: message update --------------------------
// messages = relu(messages + nei_in[src] @ W_h);  nei_out += scatter_by_tgt
__global__ __launch_bounds__(128) void gemm_msg_update(
    const float* __restrict__ nei_in,  // [N, 320]
    const bf16*  __restrict__ Wfrag,   // KC_UPD*NT fragments
    const int*   __restrict__ src,
    const int*   __restrict__ tgt,
    float* __restrict__ messages,      // [E, 320] read (C init) + write
    float* __restrict__ nei_out)       // [N, 320] pre-zeroed
{
    const int lane  = threadIdx.x & 31;
    const int wave  = threadIdx.x >> 5;
    const int mrow0 = (blockIdx.x * 4 + wave) * (16 * MT);
    const int ntSel = blockIdx.y;
    const int khalf = (lane < 16) ? 0 : 8;

    const int ncol  = lane & 15;
    const int nbase = ntSel * 16 + ncol;

    // C init: previous messages (residual add)
    v8f acc[MT][JT];
#pragma unroll
    for (int m = 0; m < MT; ++m) {
        const int rbase = mrow0 + m * 16 + ((lane < 16) ? 0 : 8);
        const float* mbase = messages + (size_t)rbase * MSG_STRIDE + nbase;
#pragma unroll
        for (int j = 0; j < JT; ++j)
#pragma unroll
            for (int r = 0; r < 8; ++r)
                acc[m][j][r] = mbase[(size_t)r * MSG_STRIDE + 80 * j];
    }

    const int srow0 = src[mrow0 + (lane & 15)];
    const int srow1 = src[mrow0 + 16 + (lane & 15)];
    const float* arow0 = nei_in + (size_t)srow0 * NEI_STRIDE + khalf;
    const float* arow1 = nei_in + (size_t)srow1 * NEI_STRIDE + khalf;
    const bf16* bbase = Wfrag + (size_t)ntSel * 512 + lane * 16;

#pragma unroll
    for (int kc = 0; kc < KC_UPD; ++kc) {
        v16bf a0 = a_frag_f32(arow0, kc * 32);
        v16bf a1 = a_frag_f32(arow1, kc * 32);
#pragma unroll
        for (int j = 0; j < JT; ++j) {
            v16bf b = b_frag(bbase, kc, j);
            acc[0][j] = __builtin_amdgcn_wmma_f32_16x16x32_bf16(
                false, a0, false, b, (short)0, acc[0][j], false, false);
            acc[1][j] = __builtin_amdgcn_wmma_f32_16x16x32_bf16(
                false, a1, false, b, (short)0, acc[1][j], false, false);
        }
    }

#pragma unroll
    for (int m = 0; m < MT; ++m) {
        const int rbase = mrow0 + m * 16 + ((lane < 16) ? 0 : 8);
        float* mbase = messages + (size_t)rbase * MSG_STRIDE + nbase;
        int t[8];
#pragma unroll
        for (int r = 0; r < 8; ++r) t[r] = tgt[rbase + r];
#pragma unroll
        for (int j = 0; j < JT; ++j) {
            int n = nbase + 80 * j;
#pragma unroll
            for (int r = 0; r < 8; ++r) {
                float v = relu(acc[m][j][r]);
                mbase[(size_t)r * MSG_STRIDE + 80 * j] = v;
                if (n < HIDDEN) atomicAdd(nei_out + (size_t)t[r] * NEI_STRIDE + n, v);
            }
        }
    }
}

// ------------------------- GEMM 3: atom output + pool ----------------------
// mol[batch[row]] += relu(A_atom @ W_o + b_o)
__global__ __launch_bounds__(128) void gemm_atom_out(
    const bf16*  __restrict__ A,       // [N, 448] bf16
    const bf16*  __restrict__ Wfrag,   // KC_OUT*NT fragments
    const float* __restrict__ bias,    // [300]
    const int*   __restrict__ batch,
    float* __restrict__ out)           // [G, 300] pre-zeroed
{
    const int lane  = threadIdx.x & 31;
    const int wave  = threadIdx.x >> 5;
    const int mrow0 = (blockIdx.x * 4 + wave) * (16 * MT);
    const int ntSel = blockIdx.y;
    const int khalf = (lane < 16) ? 0 : 8;

    const int ncol  = lane & 15;
    const int nbase = ntSel * 16 + ncol;

    v8f acc[MT][JT] = {};
#pragma unroll
    for (int j = 0; j < JT; ++j) {
        int n = nbase + 80 * j;
        float bz = (n < HIDDEN) ? bias[n] : 0.0f;
#pragma unroll
        for (int m = 0; m < MT; ++m)
#pragma unroll
            for (int r = 0; r < 8; ++r) acc[m][j][r] = bz;
    }

    const bf16* arow0 = A + (size_t)(mrow0 + (lane & 15)) * KA_OUT + khalf;
    const bf16* arow1 = arow0 + (size_t)16 * KA_OUT;
    const bf16* bbase = Wfrag + (size_t)ntSel * 512 + lane * 16;

#pragma unroll
    for (int kc = 0; kc < KC_OUT; ++kc) {
        v16bf a0 = a_frag_bf16(arow0, kc * 32);
        v16bf a1 = a_frag_bf16(arow1, kc * 32);
#pragma unroll
        for (int j = 0; j < JT; ++j) {
            v16bf b = b_frag(bbase, kc, j);
            acc[0][j] = __builtin_amdgcn_wmma_f32_16x16x32_bf16(
                false, a0, false, b, (short)0, acc[0][j], false, false);
            acc[1][j] = __builtin_amdgcn_wmma_f32_16x16x32_bf16(
                false, a1, false, b, (short)0, acc[1][j], false, false);
        }
    }

#pragma unroll
    for (int m = 0; m < MT; ++m) {
        const int rbase = mrow0 + m * 16 + ((lane < 16) ? 0 : 8);
        int g[8];
#pragma unroll
        for (int r = 0; r < 8; ++r) g[r] = batch[rbase + r];
#pragma unroll
        for (int j = 0; j < JT; ++j) {
            int n = nbase + 80 * j;
            if (n >= HIDDEN) continue;
#pragma unroll
            for (int r = 0; r < 8; ++r) {
                float v = relu(acc[m][j][r]);
                atomicAdd(out + (size_t)g[r] * HIDDEN + n, v);
            }
        }
    }
}

// ------------------------- launch ------------------------------------------

extern "C" void kernel_launch(void* const* d_in, const int* in_sizes, int n_in,
                              void* d_out, int out_size, void* d_ws, size_t ws_size,
                              hipStream_t stream) {
    (void)in_sizes; (void)n_in; (void)out_size; (void)ws_size;
    const float* x     = (const float*)d_in[0];
    const float* ea    = (const float*)d_in[1];
    const float* Wi    = (const float*)d_in[2];
    const float* Wh    = (const float*)d_in[3];
    const float* Wo    = (const float*)d_in[4];
    const float* bo    = (const float*)d_in[5];
    const int*   eidx  = (const int*)d_in[6];
    const int*   batch = (const int*)d_in[7];
    const int* src = eidx;
    const int* tgt = eidx + NUM_EDGES;
    float* out = (float*)d_out;

    char* ws = (char*)d_ws;
    size_t off = 0;
    auto alloc = [&](size_t bytes) -> char* {
        off = (off + 255) & ~(size_t)255;
        char* p = ws + off;
        off += bytes;
        return p;
    };
    bf16*  wiF      = (bf16*)alloc((size_t)KC_INIT * NT * 512 * sizeof(bf16));
    bf16*  whF      = (bf16*)alloc((size_t)KC_UPD  * NT * 512 * sizeof(bf16));
    bf16*  woF      = (bf16*)alloc((size_t)KC_OUT  * NT * 512 * sizeof(bf16));
    bf16*  Ainit    = (bf16*)alloc((size_t)NUM_EDGES * KA_INIT * sizeof(bf16));
    float* messages = (float*)alloc((size_t)NUM_EDGES * MSG_STRIDE * sizeof(float));
    float* nei0     = (float*)alloc((size_t)NUM_ATOMS * NEI_STRIDE * sizeof(float));
    float* nei1     = (float*)alloc((size_t)NUM_ATOMS * NEI_STRIDE * sizeof(float));
    bf16*  Aatom    = (bf16*)messages;   // reuse: messages dead after last update

    const int NEI_ELEMS = NUM_ATOMS * NEI_STRIDE;
    const int OUT_ELEMS = NUM_GRAPHS * HIDDEN;

    // pack weights into WMMA B-fragment layout
    prep_wfrag<<<(KC_INIT * NT * 512 + 255) / 256, 256, 0, stream>>>(Wi, wiF, ATOM_FDIM + BOND_FDIM, KC_INIT);
    prep_wfrag<<<(KC_UPD  * NT * 512 + 255) / 256, 256, 0, stream>>>(Wh, whF, HIDDEN, KC_UPD);
    prep_wfrag<<<(KC_OUT  * NT * 512 + 255) / 256, 256, 0, stream>>>(Wo, woF, ATOM_FDIM + HIDDEN, KC_OUT);

    // stage gathered/concatenated init features as bf16
    build_a_init<<<(NUM_EDGES * KA_INIT + 255) / 256, 256, 0, stream>>>(x, ea, src, Ainit);

    dim3 eg(NUM_EDGES / (64 * MT), GRIDY), ag(NUM_ATOMS / (64 * MT), GRIDY);

    // depth 1: init messages + scatter into nei0
    zero_f32<<<(NEI_ELEMS + 255) / 256, 256, 0, stream>>>(nei0, NEI_ELEMS);
    gemm_msg_init<<<eg, 128, 0, stream>>>(Ainit, wiF, tgt, messages, nei0);

    // depth 2: update + scatter into nei1
    zero_f32<<<(NEI_ELEMS + 255) / 256, 256, 0, stream>>>(nei1, NEI_ELEMS);
    gemm_msg_update<<<eg, 128, 0, stream>>>(nei0, whF, src, tgt, messages, nei1);

    // depth 3: update + scatter into nei0 (= atom_msg)
    zero_f32<<<(NEI_ELEMS + 255) / 256, 256, 0, stream>>>(nei0, NEI_ELEMS);
    gemm_msg_update<<<eg, 128, 0, stream>>>(nei1, whF, src, tgt, messages, nei0);

    // readout: concat(x, atom_msg) -> bf16, GEMM + bias + relu, pool to out
    build_a_atom<<<(NUM_ATOMS * KA_OUT + 255) / 256, 256, 0, stream>>>(x, nei0, Aatom);
    zero_f32<<<(OUT_ELEMS + 255) / 256, 256, 0, stream>>>(out, OUT_ELEMS);
    gemm_atom_out<<<ag, 128, 0, stream>>>(Aatom, woF, bo, batch, out);
}
